// GCN_76278619177596
// MI455X (gfx1250) — matrix operand order repaired and checked by reading, not compiled, not run on hardware
//
#include <hip/hip_runtime.h>
#include <hip/hip_bf16.h>

typedef __attribute__((ext_vector_type(16))) __bf16 v16bf;
typedef __attribute__((ext_vector_type(8)))  float  v8f;

// ---------------- utility kernels ----------------

__global__ void k_fill(float* __restrict__ p, float v, int n) {
    int i = blockIdx.x * blockDim.x + threadIdx.x;
    if (i < n) p[i] = v;
}

__global__ void k_degree(const int* __restrict__ col, float* __restrict__ deg, int E) {
    int e = blockIdx.x * blockDim.x + threadIdx.x;
    if (e < E) atomicAdd(&deg[col[e]], 1.0f);
}

__global__ void k_dinv(const float* __restrict__ deg, float* __restrict__ dinv, int n) {
    int i = blockIdx.x * blockDim.x + threadIdx.x;
    if (i < n) {
        float d = deg[i];
        dinv[i] = d > 0.0f ? rsqrtf(d) : 0.0f;
    }
}

// ---------------- GEMM1: xw = x @ W1  (N x 512) @ (512 x 16) ----------------
// One wave32 per 16-row tile. A: 16x32 bf16 fragment, B: 32x16 bf16 fragment,
// C/D: 16x16 f32, per CDNA5 ISA 7.12.2 layouts. v_wmma_f32_16x16x32_bf16.

__global__ void __launch_bounds__(256)
k_gemm1(const float* __restrict__ x, const float* __restrict__ W1,
        float* __restrict__ xw, int N) {
    int gtid = blockIdx.x * blockDim.x + threadIdx.x;
    int wave = gtid >> 5;
    int lane = gtid & 31;
    int base = wave * 16;
    if (base >= N) return;

    int half = lane >> 4;     // 0: lanes 0-15, 1: lanes 16-31
    int mn   = lane & 15;     // row (for A) / col (for B,C,D)
    int rowi = base + mn;
    if (rowi >= N) rowi = N - 1;               // clamp (tail tile only)
    const float* xrow = x + (size_t)rowi * 512;

    v8f acc = {};
    for (int kt = 0; kt < 16; ++kt) {
        // A fragment: elements 0..7  -> K = kt*32 + half*8 + j
        //             elements 8..15 -> K = kt*32 + 16 + half*8 + j
        int ka = kt * 32 + half * 8;
        v16bf a;
        #pragma unroll
        for (int j = 0; j < 8; ++j) {
            a[j]     = (__bf16)xrow[ka + j];
            a[j + 8] = (__bf16)xrow[ka + 16 + j];
        }
        // B fragment: column N = mn, elements 0..7 -> K = kt*32 + half*8 + j,
        //             elements 8..15 -> K = kt*32 + 16 + half*8 + j
        int kb = kt * 32 + half * 8;
        v16bf b;
        #pragma unroll
        for (int j = 0; j < 8; ++j) {
            b[j]     = (__bf16)W1[(size_t)(kb + j) * 16 + mn];
            b[j + 8] = (__bf16)W1[(size_t)(kb + 16 + j) * 16 + mn];
        }

        acc = __builtin_amdgcn_wmma_f32_16x16x32_bf16(
            false, a, false, b, (short)0, acc, false, false);
    }

    // C/D layout: VGPR r -> row M = half*8 + r, col N = mn
    float* outp = xw + (size_t)(base + half * 8) * 16 + mn;
    if (base + 16 <= N) {
        // full tile: unguarded, uniform path (always taken when N % 16 == 0)
        #pragma unroll
        for (int r = 0; r < 8; ++r)
            outp[(size_t)r * 16] = acc[r];
    } else {
        #pragma unroll
        for (int r = 0; r < 8; ++r)
            if (base + half * 8 + r < N)
                outp[(size_t)r * 16] = acc[r];
    }
}

// ---------------- GEMM2: z = h @ W2  (N x 16) @ (16 x 7) ----------------
// Single WMMA per tile: K padded 16->32 with zeros, N padded 7->16 with zeros.

__global__ void __launch_bounds__(256)
k_gemm2(const float* __restrict__ h, const float* __restrict__ W2,
        float* __restrict__ z, int N, int C) {
    int gtid = blockIdx.x * blockDim.x + threadIdx.x;
    int wave = gtid >> 5;
    int lane = gtid & 31;
    int base = wave * 16;
    if (base >= N) return;

    int half = lane >> 4;
    int mn   = lane & 15;
    int rowi = base + mn;
    if (rowi >= N) rowi = N - 1;
    const float* hrow = h + (size_t)rowi * 16;

    v16bf a = {};   // half==1 -> K = 16..31 -> stay zero
    if (half == 0) {
        #pragma unroll
        for (int j = 0; j < 16; ++j)
            a[j] = (__bf16)hrow[j];   // K = j in [0,16)
    }

    v16bf b = {};   // half==1 -> K = 16..31 -> zero pad; col >= C -> zero pad
    if (half == 0 && mn < C) {
        #pragma unroll
        for (int j = 0; j < 16; ++j)
            b[j] = (__bf16)W2[(size_t)j * C + mn];   // K = j (0..15), col = mn
    }

    v8f acc = {};
    acc = __builtin_amdgcn_wmma_f32_16x16x32_bf16(
        false, a, false, b, (short)0, acc, false, false);

    if (mn < C) {
        float* outp = z + (size_t)(base + half * 8) * C + mn;
        if (base + 16 <= N) {
            #pragma unroll
            for (int r = 0; r < 8; ++r)
                outp[(size_t)r * C] = acc[r];
        } else {
            #pragma unroll
            for (int r = 0; r < 8; ++r)
                if (base + half * 8 + r < N)
                    outp[(size_t)r * C] = acc[r];
        }
    }
}

// ---------------- edge aggregation: dst[col] += dinv[row]*dinv[col]*src[row] ----------------
// F consecutive threads handle one edge -> index loads broadcast, feature
// loads/atomics contiguous. src/dst are L2-resident (<= 6.4 MB each).

__global__ void k_agg(const int* __restrict__ row, const int* __restrict__ col,
                      const float* __restrict__ dinv, const float* __restrict__ src,
                      float* __restrict__ dst, int E, int F) {
    long long t = (long long)blockIdx.x * blockDim.x + threadIdx.x;
    long long total = (long long)E * F;
    if (t >= total) return;
    int e = (int)(t / F);
    int j = (int)(t - (long long)e * F);
    int r = row[e];
    int c = col[e];
    float nrm = dinv[r] * dinv[c];
    atomicAdd(&dst[(size_t)c * F + j], nrm * src[(size_t)r * F + j]);
}

// ---------------- finalize: dst += dinv^2 * self_src + bias; optional ReLU ----------------

__global__ void k_final(float* __restrict__ dst, const float* __restrict__ self_src,
                        const float* __restrict__ dinv, const float* __restrict__ bias,
                        int N, int F, int do_relu) {
    long long t = (long long)blockIdx.x * blockDim.x + threadIdx.x;
    long long total = (long long)N * F;
    if (t >= total) return;
    int i = (int)(t / F);
    int j = (int)(t - (long long)i * F);
    float di = dinv[i];
    float v = dst[t] + di * di * self_src[t] + bias[j];
    dst[t] = do_relu ? fmaxf(v, 0.0f) : v;
}

// ---------------- launch ----------------

extern "C" void kernel_launch(void* const* d_in, const int* in_sizes, int n_in,
                              void* d_out, int out_size, void* d_ws, size_t ws_size,
                              hipStream_t stream) {
    const float* x  = (const float*)d_in[0];
    const int*   ei = (const int*)  d_in[1];
    const float* W1 = (const float*)d_in[2];
    const float* b1 = (const float*)d_in[3];
    const float* W2 = (const float*)d_in[4];
    const float* b2 = (const float*)d_in[5];
    float* out = (float*)d_out;

    const int D = 512;
    const int H = in_sizes[3];        // 16
    const int C = in_sizes[5];        // 7
    const int N = in_sizes[0] / D;    // 100000
    const int E = in_sizes[1] / 2;    // 3200000
    const int* row = ei;              // edge_index[0] : source
    const int* col = ei + E;          // edge_index[1] : target

    float* ws   = (float*)d_ws;
    float* deg  = ws;                        // N
    float* dinv = deg  + N;                  // N
    float* xw   = dinv + N;                  // N*H
    float* h    = xw   + (size_t)N * H;      // N*H
    float* z    = h    + (size_t)N * H;      // N*C

    dim3 blk(256);
    auto nblk = [](long long n) { return dim3((unsigned)((n + 255) / 256)); };

    // init (d_out/ws are poisoned; re-init every call for determinism)
    k_fill<<<nblk(N),               blk, 0, stream>>>(deg, 1.0f, N);       // self-loop
    k_fill<<<nblk((long long)N*H),  blk, 0, stream>>>(h,   0.0f, N * H);
    k_fill<<<nblk((long long)N*C),  blk, 0, stream>>>(out, 0.0f, N * C);

    // degrees + normalization
    k_degree<<<nblk(E), blk, 0, stream>>>(col, deg, E);
    k_dinv  <<<nblk(N), blk, 0, stream>>>(deg, dinv, N);

    // layer 1
    long long gemmThreads = (long long)((N + 15) / 16) * 32;
    k_gemm1<<<nblk(gemmThreads),      blk, 0, stream>>>(x, W1, xw, N);
    k_agg  <<<nblk((long long)E * H), blk, 0, stream>>>(row, col, dinv, xw, h, E, H);
    k_final<<<nblk((long long)N * H), blk, 0, stream>>>(h, xw, dinv, b1, N, H, 1);

    // layer 2
    k_gemm2<<<nblk(gemmThreads),      blk, 0, stream>>>(h, W2, z, N, C);
    k_agg  <<<nblk((long long)E * C), blk, 0, stream>>>(row, col, dinv, z, out, E, C);
    k_final<<<nblk((long long)N * C), blk, 0, stream>>>(out, z, dinv, b2, N, C, 0);
}
